// ShiftBlock_1047972020331
// MI455X (gfx1250) — 8-bit matrix operand order repaired and checked by reading, compiled, not run on hardware
//
#include <hip/hip_runtime.h>
#include <hip/hip_bf16.h>

typedef __attribute__((ext_vector_type(8))) int   v8i;
typedef __attribute__((ext_vector_type(8))) float v8f;
typedef int vsi4 __attribute__((vector_size(16)));

#define BB 32
#define CC 256
#define HH 56
#define WW 56
#define HW (HH*WW)
#define M_TOT (BB*HW)
#define KT_PER_CONV 36
#define PW_DWORDS (3*4*KT_PER_CONV*1024)
#define A_BYTES   ((size_t)M_TOT*CC)

#ifndef __has_builtin
#define __has_builtin(x) 0
#endif
#if __has_builtin(__builtin_amdgcn_global_load_async_to_lds_b128)
#define ASYNC_LDS 1
#endif

__global__ void k_binact(const float* __restrict__ x,
                         const float* __restrict__ bias0,
                         signed char* __restrict__ a, int total) {
    int i = blockIdx.x * blockDim.x + threadIdx.x;
    if (i >= total) return;
    int xw = i % WW; int t = i / WW;
    int yh = t % HH; t /= HH;
    int c  = t % CC; int b = t / CC;
    float v = x[i] + bias0[c];
    signed char s = (v > 0.f) ? 1 : ((v < 0.f) ? -1 : 0);
    a[((((b*HH + yh)*WW + xw)) << 8) + c] = s;
    if (i < 256) a[(size_t)total + i] = 0;
}

__global__ void k_packw(const float* __restrict__ w0,
                        const float* __restrict__ w1,
                        const float* __restrict__ w2,
                        unsigned* __restrict__ pW) {
    int i = blockIdx.x * blockDim.x + threadIdx.x;
    if (i >= PW_DWORDS) return;
    int e    = i & 1;
    int lane = (i >> 1) & 31;
    int vp   = (i >> 6) & 3;
    int t    = (i >> 8) & 3;
    int kt   = (i >> 10) % KT_PER_CONV;
    int jG   = (i >> 10) / KT_PER_CONV;
    int G    = jG & 3;
    int j    = jG >> 2;
    const float* w = (j == 0) ? w0 : ((j == 1) ? w1 : w2);
    int v   = vp*2 + e;
    int n   = (G*4 + t)*16 + (lane & 15);
    int tap = kt >> 2, kh = tap / 3, kw = tap % 3;
    int c0  = (kt & 3) << 6;
    unsigned pack = 0;
    #pragma unroll
    for (int byte = 0; byte < 4; ++byte) {
        int hwk = ((v & 3) << 2) + byte + ((lane >> 4) << 4) + ((v >> 2) << 5);
        int chan = (((hwk >> 4) & 1) << 5) | (((hwk >> 5) & 1) << 4) |
                   (hwk & 15);
        int c = c0 + chan;
        float wv = w[((n*CC + c)*3 + kh)*3 + kw];
        int s = (wv > 0.f) ? 1 : ((wv < 0.f) ? -1 : 0);
        pack |= ((unsigned)(s & 0xFF)) << (8*byte);
    }
    pW[i] = pack;
}

__global__ void k_scales(const float* __restrict__ w0,
                         const float* __restrict__ w1,
                         const float* __restrict__ w2,
                         float* __restrict__ s) {
    int i = blockIdx.x * blockDim.x + threadIdx.x;
    if (i >= 3*CC) return;
    int j = i / CC, n = i % CC;
    const float* w = (j == 0) ? w0 : ((j == 1) ? w1 : w2);
    float acc = 0.f;
    for (int k = 0; k < CC*9; ++k) acc += fabsf(w[n*CC*9 + k]);
    s[i] = acc * (1.f / (CC*9.f));
}

__device__ __forceinline__ int wrap56(int v) {
    v = (v < 0)   ? v + HH : v;
    v = (v >= HH) ? v - HH : v;
    return v;
}

__global__ __launch_bounds__(256)
void k_gemm(const signed char* __restrict__ A,
            const unsigned*    __restrict__ pW,
            const float*       __restrict__ scales,
            const float*       __restrict__ xin,
            const float*       __restrict__ gamma,
            const float*       __restrict__ beta,
            const float*       __restrict__ rmean,
            const float*       __restrict__ rvar,
            const float*       __restrict__ bias1,
            const float*       __restrict__ prelu,
            const float*       __restrict__ bias2,
            float*             __restrict__ out) {
    __shared__ int4 smem4[3072];
    const unsigned* sm = (const unsigned*)smem4;

    const int lane  = threadIdx.x & 31;
    const int wave  = threadIdx.x >> 5;
    const int mtile = blockIdx.x * 8 + wave;
    const int G     = blockIdx.y;
    const int ntb   = G * 4;
    const int row   = lane & 15;
    const int half  = lane >> 4;
    const int halfoff = half << 5;
    const unsigned ZOFFH = (unsigned)A_BYTES + (unsigned)halfoff;

    const int m  = mtile*16 + row;
    const int b  = m / HW;
    const int rr = m % HW;
    const int y  = rr / WW;
    const int x  = rr % WW;
    const int b56 = b * HH;

    v8f facc[4];
    #pragma unroll
    for (int t = 0; t < 4; ++t) facc[t] = (v8f){};

    for (int j = 0; j < 3; ++j) {
        const int amt = (j == 1) ? 1 : ((j == 2) ? 3 : 0);

        int colA[3], colM[3], colP[3]; bool vw[3];
        #pragma unroll
        for (int dw = 0; dw < 3; ++dw) {
            int wq = x + dw - 1;
            vw[dw]   = (wq >= 0) & (wq < WW);
            colA[dw] = wrap56(wq);
            colM[dw] = wrap56(wq - amt);
            colP[dw] = wrap56(wq + amt);
        }

        v8i acc[4];
        #pragma unroll
        for (int t = 0; t < 4; ++t) acc[t] = (v8i){};

        for (int dh = 0; dh < 3; ++dh) {
            __syncthreads();
            const int4* src = (const int4*)pW +
                ((size_t)((j*4 + G)*KT_PER_CONV) + dh*12) * 256;
#ifdef ASYNC_LDS
            #pragma unroll
            for (int i = 0; i < 12; ++i) {
                int q = i*256 + (int)threadIdx.x;
                __builtin_amdgcn_global_load_async_to_lds_b128(
                    (vsi4*)(src + q), (vsi4*)(smem4 + q), 0, 0);
            }
            asm volatile("s_wait_asynccnt 0x0" ::: "memory");
#else
            #pragma unroll
            for (int i = 0; i < 12; ++i) {
                int q = i*256 + (int)threadIdx.x;
                smem4[q] = src[q];
            }
#endif
            __syncthreads();

            const int hq   = y + dh - 1;
            const bool vh  = (hq >= 0) & (hq < HH);
            const int rowA = (b56 + wrap56(hq)) * WW;
            const int rowM = (b56 + wrap56(hq - amt)) * WW;
            const int rowP = (b56 + wrap56(hq + amt)) * WW;

            auto aoff = [&](int idx) -> unsigned {
                const int dw = idx >> 2, g = idx & 3;
                int off;
                if      (g == 0) off = rowM + colA[dw];
                else if (g == 1) off = rowP + colA[dw];
                else if (g == 2) off = rowA + colM[dw];
                else             off = rowA + colP[dw];
                unsigned boff = ((unsigned)off << 8) + (unsigned)halfoff;
                const bool vt = vh & vw[dw];
                return vt ? boff : ZOFFH;
            };

            int4 a0c, a1c;
            {
                const unsigned bo = aoff(0);
                a0c = *(const int4*)(A + (size_t)bo);
                a1c = *(const int4*)(A + (size_t)bo + 16);
            }

            #pragma unroll
            for (int idx = 0; idx < 12; ++idx) {
                int4 a0n, a1n;
                if (idx < 11) {
                    const int gn = (idx + 1) & 3;
                    const unsigned bo = aoff(idx + 1);
                    a0n = *(const int4*)(A + (size_t)bo + gn*64);
                    a1n = *(const int4*)(A + (size_t)bo + gn*64 + 16);
                }
                v8i af;
                af[0] = a0c.x; af[1] = a0c.y; af[2] = a0c.z; af[3] = a0c.w;
                af[4] = a1c.x; af[5] = a1c.y; af[6] = a1c.z; af[7] = a1c.w;

                const int ktloc = idx;
                v8i bf[4];
                #pragma unroll
                for (int t = 0; t < 4; ++t) {
                    #pragma unroll
                    for (int vp = 0; vp < 4; ++vp) {
                        const int2 q2 = *(const int2*)
                            (sm + (ktloc*16 + t*4 + vp)*64 + lane*2);
                        bf[t][2*vp + 0] = q2.x;
                        bf[t][2*vp + 1] = q2.y;
                    }
                }
                #pragma unroll
                for (int t = 0; t < 4; ++t)
                    acc[t] = __builtin_amdgcn_wmma_i32_16x16x64_iu8(
                        true, af, true, bf[t], acc[t], false, false);

                if (idx < 11) { a0c = a0n; a1c = a1n; }
            }
        }
        #pragma unroll
        for (int t = 0; t < 4; ++t) {
            float s = scales[j*CC + (ntb + t)*16 + row];
            #pragma unroll
            for (int v = 0; v < 8; ++v) facc[t][v] += s * (float)acc[t][v];
        }
    }

    const int m0  = mtile*16 + (half << 3);
    const int bb  = m0 / HW;
    const int rr2 = m0 % HW;
    const int yy  = rr2 / WW;
    const int xx  = rr2 % WW;
    #pragma unroll
    for (int t = 0; t < 4; ++t) {
        const int n  = (ntb + t)*16 + row;
        const float k1 = gamma[n] * rsqrtf(rvar[n] + 1e-5f);
        const float rm = rmean[n], be = beta[n];
        const float b1 = bias1[n], pa = prelu[n], b2 = bias2[n];
        const int idx0 = ((bb*CC + n)*HH + yy)*WW + xx;
        const float4 r0 = *(const float4*)(xin + idx0);
        const float4 r1 = *(const float4*)(xin + idx0 + 4);
        const float res[8] = {r0.x, r0.y, r0.z, r0.w, r1.x, r1.y, r1.z, r1.w};
        float o[8];
        #pragma unroll
        for (int v = 0; v < 8; ++v) {
            float val = (facc[t][v] - rm) * k1 + be + res[v] + b1;
            val = (val > 0.f) ? val : pa * val;
            o[v] = val + b2;
        }
        *(float4*)(out + idx0)     = make_float4(o[0], o[1], o[2], o[3]);
        *(float4*)(out + idx0 + 4) = make_float4(o[4], o[5], o[6], o[7]);
    }
}

extern "C" void kernel_launch(void* const* d_in, const int* in_sizes, int n_in,
                              void* d_out, int out_size, void* d_ws, size_t ws_size,
                              hipStream_t stream) {
    const float* x     = (const float*)d_in[0];
    const float* bias0 = (const float*)d_in[1];
    const float* w0    = (const float*)d_in[2];
    const float* w1    = (const float*)d_in[3];
    const float* w2    = (const float*)d_in[4];
    const float* gamma = (const float*)d_in[5];
    const float* beta  = (const float*)d_in[6];
    const float* rmean = (const float*)d_in[7];
    const float* rvar  = (const float*)d_in[8];
    const float* bias1 = (const float*)d_in[9];
    const float* prelu = (const float*)d_in[10];
    const float* bias2 = (const float*)d_in[11];
    float* out = (float*)d_out;

    char* ws = (char*)d_ws;
    signed char* a_i8  = (signed char*)ws;
    unsigned*    pW    = (unsigned*)(ws + A_BYTES + 256);
    float*       scl   = (float*)(ws + A_BYTES + 256 + (size_t)PW_DWORDS*4);

    const int total = BB*CC*HH*WW;
    k_binact<<<total/256, 256, 0, stream>>>(x, bias0, a_i8, total);
    k_packw <<<PW_DWORDS/256, 256, 0, stream>>>(w0, w1, w2, pW);
    k_scales<<<3, 256, 0, stream>>>(w0, w1, w2, scl);

    dim3 grid(M_TOT/16/8, CC/64, 1);
    k_gemm<<<grid, 256, 0, stream>>>(a_i8, pW, scl, x,
                                     gamma, beta, rmean, rvar,
                                     bias1, prelu, bias2, out);
}
